// EncoderLayer_78374563217960
// MI455X (gfx1250) — matrix operand order repaired and checked by reading, not compiled, not run on hardware
//
#include <hip/hip_runtime.h>
#include <hip/hip_bf16.h>

typedef __attribute__((ext_vector_type(4)))  _Float16 v4h;
typedef __attribute__((ext_vector_type(8)))  _Float16 v8h;
typedef __attribute__((ext_vector_type(16))) _Float16 v16h;
typedef __attribute__((ext_vector_type(8)))  float    v8f;
typedef __attribute__((ext_vector_type(4)))  float    v4f;

#define DEV __device__ __forceinline__

constexpr int CD    = 512;          // d_model
constexpr int NH    = 8;            // heads
constexpr int HD    = 64;           // head dim
constexpr int NB    = 4;            // batch
constexpr int SEQ   = 2048;         // seq len
constexpr int MROWS = NB * SEQ;     // 8192 total rows

// ---------------------------------------------------------------- WMMA core
DEV v8f wmma16(v16h a, v16h b, v8f c) {
  // D = A(16x32 f16) * B(32x16 f16) + C(16x16 f32)
  return __builtin_amdgcn_wmma_f32_16x16x32_f16(false, a, false, b,
                                                (short)0, c, false, false);
}

// A fragment (16x32, row-major, stride ld): lane r=lane&15 holds row r.
// half j of v16h: K = (j>>3)*16 + (lane>>4)*8 + (j&7)  -> two contiguous v8h.
DEV v16h load_fragA(const _Float16* p, int ld, int lane) {
  const int r = lane & 15, kq = (lane >> 4) & 1;
  const _Float16* q = p + (size_t)r * ld + kq * 8;
  v8h lo = *(const v8h*)q;
  v8h hi = *(const v8h*)(q + 16);
  v16h out;
#pragma unroll
  for (int i = 0; i < 8; ++i) { out[i] = lo[i]; out[i + 8] = hi[i]; }
  return out;
}

// B fragment (32x16 K-major view): source is row-major [16 rows n x K], i.e. a
// weight matrix W[out,in] tile; lane n=lane&15 reads K-contiguous halves.
// half j: K = (j>>3)*16 + (lane>>4)*8 + (j&7).
DEV v16h load_fragB(const _Float16* p, int ld, int lane) {
  const int n = lane & 15, kg = (lane >> 4) & 1;
  const _Float16* q = p + (size_t)n * ld + kg * 8;
  v8h lo = *(const v8h*)q;
  v8h hi = *(const v8h*)(q + 16);
  v16h out;
#pragma unroll
  for (int i = 0; i < 8; ++i) { out[i] = lo[i]; out[i + 8] = hi[i]; }
  return out;
}

// reductions across a 16-lane half (C-tile rows live per half)
DEV float halfmax16(float v) {
#pragma unroll
  for (int off = 1; off < 16; off <<= 1) v = fmaxf(v, __shfl_xor(v, off, 32));
  return v;
}
DEV float halfsum16(float v) {
#pragma unroll
  for (int off = 1; off < 16; off <<= 1) v += __shfl_xor(v, off, 32);
  return v;
}

// ---------------------------------------------------------------- helpers
__global__ void f32_to_f16_kernel(const float* __restrict__ x,
                                  _Float16* __restrict__ y, int n) {
  int i = blockIdx.x * blockDim.x + threadIdx.x;
  if (i < n) y[i] = (_Float16)x[i];
}

// LayerNorm over 512 cols, one wave per row, f16 output.
__global__ __launch_bounds__(256) void ln_f16_kernel(
    const float* __restrict__ X, const float* __restrict__ g,
    const float* __restrict__ b, _Float16* __restrict__ Y) {
  const int lane = threadIdx.x & 31;
  const int row  = blockIdx.x * 8 + (threadIdx.x >> 5);
  const float* xr = X + (size_t)row * CD;
  v4f xv[4];
  float s = 0.f, s2 = 0.f;
#pragma unroll
  for (int c = 0; c < 4; ++c) {
    xv[c] = *(const v4f*)(xr + lane * 4 + c * 128);
#pragma unroll
    for (int e = 0; e < 4; ++e) { s += xv[c][e]; s2 += xv[c][e] * xv[c][e]; }
  }
#pragma unroll
  for (int off = 1; off < 32; off <<= 1) {
    s  += __shfl_xor(s,  off, 32);
    s2 += __shfl_xor(s2, off, 32);
  }
  const float mu = s * (1.f / CD);
  const float rs = rsqrtf(s2 * (1.f / CD) - mu * mu + 1e-5f);
#pragma unroll
  for (int c = 0; c < 4; ++c) {
    const int col = lane * 4 + c * 128;
    v4f gv = *(const v4f*)(g + col);
    v4f bv = *(const v4f*)(b + col);
    v4h yv;
#pragma unroll
    for (int e = 0; e < 4; ++e)
      yv[e] = (_Float16)((xv[c][e] - mu) * rs * gv[e] + bv[e]);
    *(v4h*)(Y + (size_t)row * CD + col) = yv;
  }
}

// ------------------------------------------------- generic WMMA GEMM: X @ W^T
// A: [M,K] f16 row-major, W: [N,K] f16 row-major (torch Linear layout).
// 256 threads = 8 waves (2M x 4N), wave tile 64x64, block tile 128x256.
// EPI: 0 = f16 store, 1 = f16 ReLU store, 2 = f32 store of (R + A@W^T).
template <int EPI>
__global__ __launch_bounds__(256) void gemm_wmma_kernel(
    const _Float16* __restrict__ A, const _Float16* __restrict__ W,
    int M, int N, int K, _Float16* __restrict__ Ch, float* Cf, const float* R) {
  const int lane = threadIdx.x & 31;
  const int wid  = threadIdx.x >> 5;
  const int m0 = blockIdx.y * 128 + (wid >> 2) * 64;
  const int n0 = blockIdx.x * 256 + (wid & 3) * 64;

  const v8f vzero = {};
  v8f acc[4][4];
#pragma unroll
  for (int i = 0; i < 4; ++i)
#pragma unroll
    for (int j = 0; j < 4; ++j) acc[i][j] = vzero;

  for (int k0 = 0; k0 < K; k0 += 32) {
    if (k0 + 32 < K) {  // L2-feeding prefetch of next K slab
      __builtin_prefetch(A + (size_t)(m0 + lane) * K + k0 + 32, 0, 1);
      __builtin_prefetch(W + (size_t)(n0 + lane) * K + k0 + 32, 0, 1);
    }
    v16h a[4], b[4];
#pragma unroll
    for (int i = 0; i < 4; ++i)
      a[i] = load_fragA(A + (size_t)(m0 + i * 16) * K + k0, K, lane);
#pragma unroll
    for (int j = 0; j < 4; ++j)
      b[j] = load_fragB(W + (size_t)(n0 + j * 16) * K + k0, K, lane);
#pragma unroll
    for (int i = 0; i < 4; ++i)
#pragma unroll
      for (int j = 0; j < 4; ++j) acc[i][j] = wmma16(a[i], b[j], acc[i][j]);
  }

  const int r = lane & 15, hi = lane >> 4;
#pragma unroll
  for (int i = 0; i < 4; ++i)
#pragma unroll
    for (int j = 0; j < 4; ++j)
#pragma unroll
      for (int v = 0; v < 8; ++v) {
        const size_t row = (size_t)(m0 + i * 16 + v + 8 * hi);
        const size_t col = (size_t)(n0 + j * 16 + r);
        const float val = acc[i][j][v];
        if (EPI == 0)
          Ch[row * N + col] = (_Float16)val;
        else if (EPI == 1)
          Ch[row * N + col] = (_Float16)fmaxf(val, 0.f);
        else
          Cf[row * N + col] = R[row * N + col] + val;
      }
}

// ------------------------------------------------- flash attention (mask==1)
// Q/K/V/O: [NB*SEQ, 512] f16; head h occupies cols [h*64, h*64+64).
// Block: 128 threads = 4 waves; wave owns 16 query rows; s-chunks of 64.
__global__ __launch_bounds__(128) void flash_attn_kernel(
    const _Float16* __restrict__ Q, const _Float16* __restrict__ K,
    const _Float16* __restrict__ V, _Float16* __restrict__ O) {
  const int h    = blockIdx.y;
  const int bat  = blockIdx.z;
  const int wid  = threadIdx.x >> 5;
  const int lane = threadIdx.x & 31;
  const int q0   = blockIdx.x * 64 + wid * 16;
  const int base = bat * SEQ;
  const int hc   = h * HD;
  const int r = lane & 15, half = lane >> 4;

  __shared__ _Float16 vt[HD][72];        // V chunk, transposed [d][s]
  __shared__ _Float16 pbuf[4][16][72];   // per-wave P tile, A-layout staging

  const v8f vzero = {};
  v8f accO[4];
#pragma unroll
  for (int j = 0; j < 4; ++j) accO[j] = vzero;
  float mrow[8], lrow[8];
#pragma unroll
  for (int v = 0; v < 8; ++v) { mrow[v] = -3.0e38f; lrow[v] = 0.f; }

  for (int s0 = 0; s0 < SEQ; s0 += 64) {
    // cooperative transpose staging of the V chunk: 16B global loads,
    // transpose handled on the LDS-store side. 64x64 halves = 512 v8h chunks.
#pragma unroll
    for (int it = 0; it < 4; ++it) {
      const int c  = threadIdx.x + it * 128;
      const int s  = c >> 3;          // source row (seq position)
      const int d0 = (c & 7) * 8;     // first of 8 head-dim elements
      v8h vv = *(const v8h*)(V + (size_t)(base + s0 + s) * CD + hc + d0);
#pragma unroll
      for (int e = 0; e < 8; ++e) vt[d0 + e][s] = vv[e];
    }
    __syncthreads();

    // scores S = Q(16x64) * K^T(64x64), fp32 accum
    v8f accS[4];
#pragma unroll
    for (int j = 0; j < 4; ++j) accS[j] = vzero;
#pragma unroll
    for (int dk = 0; dk < HD; dk += 32) {
      v16h aq = load_fragA(Q + (size_t)(base + q0) * CD + hc + dk, CD, lane);
#pragma unroll
      for (int j = 0; j < 4; ++j) {
        v16h bk = load_fragB(K + (size_t)(base + s0 + j * 16) * CD + hc + dk,
                             CD, lane);
        accS[j] = wmma16(aq, bk, accS[j]);
      }
    }

    // online softmax (temp = 1/sqrt(64) = 0.125); rows live per (v, lane-half)
    float pj[4][8];
#pragma unroll
    for (int v = 0; v < 8; ++v) {
      float mx = -3.0e38f;
#pragma unroll
      for (int j = 0; j < 4; ++j) mx = fmaxf(mx, accS[j][v]);
      mx = halfmax16(mx) * 0.125f;
      const float mnew  = fmaxf(mrow[v], mx);
      const float alpha = __expf(mrow[v] - mnew);
      mrow[v] = mnew;
      float rsum = 0.f;
#pragma unroll
      for (int j = 0; j < 4; ++j) {
        const float p = __expf(accS[j][v] * 0.125f - mnew);
        pj[j][v] = p;
        rsum += p;
      }
      rsum = halfsum16(rsum);
      lrow[v] = lrow[v] * alpha + rsum;
#pragma unroll
      for (int jd = 0; jd < 4; ++jd) accO[jd][v] *= alpha;
    }

    // C-layout -> A-layout via LDS staging
#pragma unroll
    for (int j = 0; j < 4; ++j)
#pragma unroll
      for (int v = 0; v < 8; ++v)
        pbuf[wid][v + 8 * half][j * 16 + r] = (_Float16)pj[j][v];
    // Architectural + compiler ordering barrier: the staging stores above must
    // not be reordered against the fragment loads below (lanes read P values
    // written by other lanes of the same wave).
    __syncthreads();

    // O += P(16x64) * V(64x64)
#pragma unroll
    for (int jd = 0; jd < 4; ++jd)
#pragma unroll
      for (int kk = 0; kk < 64; kk += 32) {
        v16h ap = load_fragA(&pbuf[wid][0][kk], 72, lane);
        v16h bv = load_fragB(&vt[jd * 16][kk], 72, lane);
        accO[jd] = wmma16(ap, bv, accO[jd]);
      }
    __syncthreads();
  }

  // normalize and store (head-interleaved back into [row, 512])
#pragma unroll
  for (int jd = 0; jd < 4; ++jd)
#pragma unroll
    for (int v = 0; v < 8; ++v) {
      const float val = accO[jd][v] / lrow[v];
      O[(size_t)(base + q0 + v + 8 * half) * CD + hc + jd * 16 + r] =
          (_Float16)val;
    }
}

// ---------------------------------------------------------------- launcher
extern "C" void kernel_launch(void* const* d_in, const int* in_sizes, int n_in,
                              void* d_out, int out_size, void* d_ws,
                              size_t ws_size, hipStream_t stream) {
  (void)in_sizes; (void)n_in; (void)out_size; (void)ws_size;
  const float* x  = (const float*)d_in[0];
  // d_in[1] = x_mask: all true in this problem -> dense softmax path
  const float* Wq = (const float*)d_in[2];
  const float* Wk = (const float*)d_in[3];
  const float* Wv = (const float*)d_in[4];
  const float* Wm = (const float*)d_in[5];
  const float* W1 = (const float*)d_in[6];
  const float* W2 = (const float*)d_in[7];
  const float* g1 = (const float*)d_in[8];
  const float* b1 = (const float*)d_in[9];
  const float* g2 = (const float*)d_in[10];
  const float* b2 = (const float*)d_in[11];
  float* out = (float*)d_out;

  char*  ws  = (char*)d_ws;
  size_t off = 0;
  auto alloc = [&](size_t bytes) -> char* {
    char* p = ws + off;
    off += (bytes + 255) & ~(size_t)255;
    return p;
  };
  _Float16* xn  = (_Float16*)alloc((size_t)MROWS * CD * 2);
  _Float16* qh  = (_Float16*)alloc((size_t)MROWS * CD * 2);
  _Float16* kh  = (_Float16*)alloc((size_t)MROWS * CD * 2);
  _Float16* vh  = (_Float16*)alloc((size_t)MROWS * CD * 2);
  _Float16* att = (_Float16*)alloc((size_t)MROWS * CD * 2);
  _Float16* mn  = (_Float16*)alloc((size_t)MROWS * CD * 2);
  _Float16* h1  = (_Float16*)alloc((size_t)MROWS * 2 * CD * 2);
  _Float16* wqh = (_Float16*)alloc((size_t)CD * CD * 2);
  _Float16* wkh = (_Float16*)alloc((size_t)CD * CD * 2);
  _Float16* wvh = (_Float16*)alloc((size_t)CD * CD * 2);
  _Float16* wmh = (_Float16*)alloc((size_t)CD * CD * 2);
  _Float16* w1h = (_Float16*)alloc((size_t)2 * CD * CD * 2);
  _Float16* w2h = (_Float16*)alloc((size_t)2 * CD * CD * 2);

  // weight conversion (f32 -> f16)
  {
    const int nw = CD * CD;
    f32_to_f16_kernel<<<(nw + 255) / 256, 256, 0, stream>>>(Wq, wqh, nw);
    f32_to_f16_kernel<<<(nw + 255) / 256, 256, 0, stream>>>(Wk, wkh, nw);
    f32_to_f16_kernel<<<(nw + 255) / 256, 256, 0, stream>>>(Wv, wvh, nw);
    f32_to_f16_kernel<<<(nw + 255) / 256, 256, 0, stream>>>(Wm, wmh, nw);
    f32_to_f16_kernel<<<(2 * nw + 255) / 256, 256, 0, stream>>>(W1, w1h, 2 * nw);
    f32_to_f16_kernel<<<(2 * nw + 255) / 256, 256, 0, stream>>>(W2, w2h, 2 * nw);
  }

  // LN1
  ln_f16_kernel<<<MROWS / 8, 256, 0, stream>>>(x, g1, b1, xn);

  // Q/K/V projections
  dim3 gP(CD / 256, MROWS / 128);
  gemm_wmma_kernel<0><<<gP, 256, 0, stream>>>(xn, wqh, MROWS, CD, CD, qh,
                                              nullptr, nullptr);
  gemm_wmma_kernel<0><<<gP, 256, 0, stream>>>(xn, wkh, MROWS, CD, CD, kh,
                                              nullptr, nullptr);
  gemm_wmma_kernel<0><<<gP, 256, 0, stream>>>(xn, wvh, MROWS, CD, CD, vh,
                                              nullptr, nullptr);

  // attention
  flash_attn_kernel<<<dim3(SEQ / 64, NH, NB), 128, 0, stream>>>(qh, kh, vh, att);

  // merge projection + residual -> d_out (fp32)
  gemm_wmma_kernel<2><<<gP, 256, 0, stream>>>(att, wmh, MROWS, CD, CD, nullptr,
                                              out, x);

  // LN2
  ln_f16_kernel<<<MROWS / 8, 256, 0, stream>>>(out, g2, b2, mn);

  // MLP
  dim3 g1d(2 * CD / 256, MROWS / 128);
  gemm_wmma_kernel<1><<<g1d, 256, 0, stream>>>(mn, w1h, MROWS, 2 * CD, CD, h1,
                                               nullptr, nullptr);
  gemm_wmma_kernel<2><<<gP, 256, 0, stream>>>(h1, w2h, MROWS, CD, 2 * CD,
                                              nullptr, out, out);
}
